// Informer_71588514890199
// MI455X (gfx1250) — matrix operand order repaired and checked by reading, not compiled, not run on hardware
//
#include <hip/hip_runtime.h>
#include <hip/hip_bf16.h>
#include <math.h>

// ---------------- constants (Informer config) ----------------
#define BB      4
#define LL      2048
#define ENCIN   32
#define DM      256
#define HH      8
#define HD      2048          // H * DM
#define DFF     1024
#define SK      40            // SAMPLE_K
#define NTOP    5
#define ELAY    2

typedef __bf16 bf16;
typedef __attribute__((ext_vector_type(16))) __bf16 v16bf;
typedef __attribute__((ext_vector_type(8)))  float  v8f;
typedef int inf_v4i __attribute__((vector_size(16)));

#if defined(__AMDGCN__) && __has_builtin(__builtin_amdgcn_global_load_async_to_lds_b128)
#define USE_ASYNC_LDS 1
#else
#define USE_ASYNC_LDS 0
#endif

// =================================================================
// elementwise f32 -> bf16
// =================================================================
__global__ __launch_bounds__(256) void inf_cvt_bf16(const float* __restrict__ in,
                                                    bf16* __restrict__ out, int n) {
  int i = blockIdx.x * 256 + threadIdx.x;
  if (i < n) out[i] = (bf16)in[i];
}

// =================================================================
// W [K,N] f32  ->  Wt [N,K] bf16   (tiled transpose via LDS)
// =================================================================
__global__ __launch_bounds__(256) void inf_transpose_cvt(const float* __restrict__ W,
                                                         bf16* __restrict__ Wt,
                                                         int K, int N) {
  __shared__ float t[32][33];
  int tx = threadIdx.x, ty = threadIdx.y;
  int n = blockIdx.x * 32 + tx;
#pragma unroll
  for (int j = 0; j < 4; ++j) {
    int k = blockIdx.y * 32 + ty + j * 8;
    t[ty + j * 8][tx] = (k < K && n < N) ? W[(size_t)k * N + n] : 0.f;
  }
  __syncthreads();
  int k2 = blockIdx.y * 32 + tx;
#pragma unroll
  for (int j = 0; j < 4; ++j) {
    int n2 = blockIdx.x * 32 + ty + j * 8;
    if (n2 < N && k2 < K) Wt[(size_t)n2 * K + k2] = (bf16)t[tx][ty + j * 8];
  }
}

// =================================================================
// WMMA GEMM:  C[M,N] = A[M,K](bf16,row) @ Bt[N,K](bf16,row)^T + bias
// block tile 128x128, 8 waves (wave32), each wave 32x64 = 2x4 WMMA units.
// Double-buffered LDS (padded stride 40 halfwords), filled with
// GLOBAL_LOAD_ASYNC_TO_LDS_B128 (no VGPR staging), pipelined with
// s_wait_asynccnt + workgroup barrier.
// =================================================================
union InfFrag { uint4 u[2]; v16bf v; };

__device__ __forceinline__ void inf_async16(const void* g, void* l) {
#if USE_ASYNC_LDS
  __builtin_amdgcn_global_load_async_to_lds_b128(
      (__attribute__((address_space(1))) inf_v4i*)(void*)(const_cast<void*>(g)),
      (__attribute__((address_space(3))) inf_v4i*)(l), 0, 0);
#else
  (void)g; (void)l;
#endif
}

__device__ __forceinline__ void inf_wait_async() {
#if USE_ASYNC_LDS
#if __has_builtin(__builtin_amdgcn_s_wait_asynccnt)
  __builtin_amdgcn_s_wait_asynccnt(0);
#else
  asm volatile("s_wait_asynccnt 0x0" ::: "memory");
#endif
#endif
}

template <int ACT, bool WF, bool WB>
__global__ __launch_bounds__(256) void inf_gemm_bf16(
    const bf16* __restrict__ A, const bf16* __restrict__ Bt,
    const float* __restrict__ bias,
    float* __restrict__ Cf, bf16* __restrict__ Cb,
    int M, int N, int K) {
  __shared__ __align__(16) bf16 As[2][128 * 40];
  __shared__ __align__(16) bf16 Bs[2][128 * 40];

  const int tid  = threadIdx.x;
  const int lane = tid & 31;
  const int wid  = tid >> 5;
  const int wrow = wid >> 1;        // 0..3  -> 32-row strip
  const int wcol = wid & 1;         // 0..1  -> 64-col strip
  const int m0 = blockIdx.y * 128;
  const int n0 = blockIdx.x * 128;
  const int half = lane >> 4;
  const int l15  = lane & 15;

  // per-thread tile-fill mapping: 2 x 16B chunks of A, 2 of B
  const int idx0 = tid, idx1 = tid + 256;
  const int r0 = idx0 >> 2, kk0 = (idx0 & 3) << 3;
  const int r1 = idx1 >> 2, kk1 = (idx1 & 3) << 3;
  const bf16* Ap0 = A  + (size_t)(m0 + r0) * K + kk0;
  const bf16* Ap1 = A  + (size_t)(m0 + r1) * K + kk1;
  const bf16* Bp0 = Bt + (size_t)(n0 + r0) * K + kk0;
  const bf16* Bp1 = Bt + (size_t)(n0 + r1) * K + kk1;
  const int ofs0 = r0 * 40 + kk0;
  const int ofs1 = r1 * 40 + kk1;

  v8f acc[2][4];
#pragma unroll
  for (int i = 0; i < 2; ++i)
#pragma unroll
    for (int j = 0; j < 4; ++j) acc[i][j] = (v8f){};

  const int KT = K >> 5;

  auto issue_tile = [&](int buf, int kt2) {
    const size_t ko = (size_t)kt2 << 5;
#if USE_ASYNC_LDS
    inf_async16(Ap0 + ko, &As[buf][ofs0]);
    inf_async16(Ap1 + ko, &As[buf][ofs1]);
    inf_async16(Bp0 + ko, &Bs[buf][ofs0]);
    inf_async16(Bp1 + ko, &Bs[buf][ofs1]);
#else
    uint4 a0 = *(const uint4*)(Ap0 + ko);
    uint4 a1 = *(const uint4*)(Ap1 + ko);
    uint4 b0 = *(const uint4*)(Bp0 + ko);
    uint4 b1 = *(const uint4*)(Bp1 + ko);
    *(uint4*)&As[buf][ofs0] = a0;
    *(uint4*)&As[buf][ofs1] = a1;
    *(uint4*)&Bs[buf][ofs0] = b0;
    *(uint4*)&Bs[buf][ofs1] = b1;
#endif
  };

  issue_tile(0, 0);
  inf_wait_async();
  __syncthreads();

  for (int kt = 0; kt < KT; ++kt) {
    const int cur = kt & 1;
    const bool pre = (kt + 1 < KT);
    if (pre) issue_tile(cur ^ 1, kt + 1);   // async fill of the idle buffer

    // ---- fragments from LDS ----
    InfFrag fa[2], fb[4];
#pragma unroll
    for (int i = 0; i < 2; ++i) {
      int r = wrow * 32 + i * 16 + l15;
      fa[i].u[0] = *(const uint4*)&As[cur][r * 40 + half * 8];
      fa[i].u[1] = *(const uint4*)&As[cur][r * 40 + 16 + half * 8];
    }
#pragma unroll
    for (int j = 0; j < 4; ++j) {
      int c = wcol * 64 + j * 16 + l15;
      fb[j].u[0] = *(const uint4*)&Bs[cur][c * 40 + half * 8];
      fb[j].u[1] = *(const uint4*)&Bs[cur][c * 40 + 16 + half * 8];
    }

#pragma unroll
    for (int i = 0; i < 2; ++i)
#pragma unroll
      for (int j = 0; j < 4; ++j)
        acc[i][j] = __builtin_amdgcn_wmma_f32_16x16x32_bf16(
            false, fa[i].v, false, fb[j].v, (short)0, acc[i][j], false, false);

    if (pre) {
      inf_wait_async();
      __syncthreads();
    }
  }

  // ---- epilogue (bias always present; straight-line stores) ----
#pragma unroll
  for (int i = 0; i < 2; ++i) {
#pragma unroll
    for (int j = 0; j < 4; ++j) {
      int col = n0 + wcol * 64 + j * 16 + l15;
      float bc = bias[col];
#pragma unroll
      for (int g = 0; g < 8; ++g) {
        int row = m0 + wrow * 32 + i * 16 + g + half * 8;
        float v = acc[i][j][g] + bc;
        if (ACT == 1) v = 0.5f * v * (1.0f + erff(v * 0.70710678118654752f));
        if (WF) Cf[(size_t)row * N + col] = v;
        if (WB) Cb[(size_t)row * N + col] = (bf16)v;
      }
    }
  }
}

// =================================================================
// helpers
// =================================================================
__device__ __forceinline__ void inf_load8(const bf16* p, float* o) {
  uint4 u = *(const uint4*)p;
  const bf16* bp = (const bf16*)&u;
#pragma unroll
  for (int i = 0; i < 8; ++i) o[i] = (float)bp[i];
}

// =================================================================
// sparsity measure M[b,h,l] = max_u s_u - (1/L) * sum_u s_u
// one wave per (b,h,l); q,k bf16 [B*L, HD], head slice h*DM
// =================================================================
__global__ __launch_bounds__(256) void inf_sample_m(
    const bf16* __restrict__ q, const bf16* __restrict__ k,
    const int* __restrict__ samp, float* __restrict__ Mb) {
  int lane = threadIdx.x & 31;
  int wid  = threadIdx.x >> 5;
  int g    = blockIdx.x * 8 + wid;          // 0 .. B*H*L-1
  int l    = g & (LL - 1);
  int bh   = g >> 11;                       // L = 2048
  int b    = bh >> 3;
  int h    = bh & 7;

  const bf16* qrow = q + ((size_t)(b * LL + l)) * HD + h * DM + lane * 8;
  float qv[8];
  inf_load8(qrow, qv);

  float mx = -1e30f, sm = 0.f;
  for (int u = 0; u < SK; ++u) {
    int j = samp[l * SK + u];
    const bf16* krow = k + ((size_t)(b * LL + j)) * HD + h * DM + lane * 8;
    float kv[8];
    inf_load8(krow, kv);
    float p = 0.f;
#pragma unroll
    for (int i = 0; i < 8; ++i) p += qv[i] * kv[i];
#pragma unroll
    for (int off = 16; off > 0; off >>= 1) p += __shfl_xor(p, off, 32);
    mx = fmaxf(mx, p);
    sm += p;
  }
  if (lane == 0) Mb[(size_t)bh * LL + l] = mx - sm / (float)LL;
}

// =================================================================
// top-5 of M[b,h,:] -> mtop[b,h,5]; one block per (b,h)
// =================================================================
__global__ __launch_bounds__(256) void inf_topk(const float* __restrict__ Mb,
                                                int* __restrict__ mtop) {
  __shared__ float sv[LL];
  __shared__ float rv[256];
  __shared__ int   ri[256];
  int bh = blockIdx.x, t = threadIdx.x;
#pragma unroll
  for (int i = 0; i < 8; ++i) sv[t + i * 256] = Mb[(size_t)bh * LL + t + i * 256];
  __syncthreads();
  for (int r = 0; r < NTOP; ++r) {
    float best = -1e30f; int bi = 0;
#pragma unroll
    for (int i = 0; i < 8; ++i) {
      float v = sv[t + i * 256];
      if (v > best) { best = v; bi = t + i * 256; }
    }
    rv[t] = best; ri[t] = bi;
    __syncthreads();
    for (int s = 128; s > 0; s >>= 1) {
      if (t < s && rv[t + s] > rv[t]) { rv[t] = rv[t + s]; ri[t] = ri[t + s]; }
      __syncthreads();
    }
    if (t == 0) { mtop[bh * NTOP + r] = ri[0]; sv[ri[0]] = -1e30f; }
    __syncthreads();
  }
}

// =================================================================
// top-query context: softmax(q_m K^T) V, one block per (b,h,n)
// =================================================================
__global__ __launch_bounds__(256) void inf_ctx(
    const bf16* __restrict__ q, const bf16* __restrict__ k,
    const bf16* __restrict__ v, const int* __restrict__ mtop,
    float* __restrict__ ctx) {
  __shared__ float sc[LL];
  __shared__ float red[256];
  int n = blockIdx.x, h = blockIdx.y, b = blockIdx.z;
  int t = threadIdx.x, lane = t & 31, wid = t >> 5;
  int bh = b * HH + h;
  int m  = mtop[bh * NTOP + n];

  float qv[8];
  inf_load8(q + ((size_t)(b * LL + m)) * HD + h * DM + lane * 8, qv);

  for (int it = 0; it < 256; ++it) {
    int j = it * 8 + wid;
    float kv[8];
    inf_load8(k + ((size_t)(b * LL + j)) * HD + h * DM + lane * 8, kv);
    float p = 0.f;
#pragma unroll
    for (int i = 0; i < 8; ++i) p += qv[i] * kv[i];
#pragma unroll
    for (int off = 16; off > 0; off >>= 1) p += __shfl_xor(p, off, 32);
    if (lane == 0) sc[j] = p;
  }
  __syncthreads();

  // softmax over 2048
  float lm = -1e30f;
#pragma unroll
  for (int i = 0; i < 8; ++i) lm = fmaxf(lm, sc[t + i * 256]);
  red[t] = lm; __syncthreads();
  for (int s = 128; s > 0; s >>= 1) { if (t < s) red[t] = fmaxf(red[t], red[t + s]); __syncthreads(); }
  float mx = red[0]; __syncthreads();
  float ls = 0.f;
#pragma unroll
  for (int i = 0; i < 8; ++i) { float e = expf(sc[t + i * 256] - mx); sc[t + i * 256] = e; ls += e; }
  red[t] = ls; __syncthreads();
  for (int s = 128; s > 0; s >>= 1) { if (t < s) red[t] += red[t + s]; __syncthreads(); }
  float inv = 1.f / red[0]; __syncthreads();
#pragma unroll
  for (int i = 0; i < 8; ++i) sc[t + i * 256] *= inv;
  __syncthreads();

  // ctx[d] = sum_j p_j * v[j][d]   (d = t, coalesced v reads)
  float acc = 0.f;
  for (int j = 0; j < LL; ++j)
    acc += sc[j] * (float)v[((size_t)(b * LL + j)) * HD + h * DM + t];
  ctx[((size_t)bh * NTOP + n) * DM + t] = acc;
}

// =================================================================
// zero fill (16B granules) + scatter ctx rows into a_flat
// =================================================================
__global__ __launch_bounds__(256) void inf_zero16(uint4* __restrict__ p, int n16) {
  int i = blockIdx.x * 256 + threadIdx.x;
  if (i < n16) { uint4 z = {0u, 0u, 0u, 0u}; p[i] = z; }
}

__global__ __launch_bounds__(256) void inf_scatter(
    const float* __restrict__ ctx, const int* __restrict__ mtop,
    bf16* __restrict__ aflat) {
  int n = blockIdx.x, h = blockIdx.y, b = blockIdx.z, t = threadIdx.x;
  int bh = b * HH + h;
  int m  = mtop[bh * NTOP + n];
  aflat[((size_t)(b * LL + m)) * HD + h * DM + t] =
      (bf16)ctx[((size_t)bh * NTOP + n) * DM + t];
}

// =================================================================
// x = LN(x + dx) ; writes f32 (in place ok) + bf16 copy. 1 block / row.
// =================================================================
__global__ __launch_bounds__(256) void inf_ln_res(
    const float* __restrict__ x, const float* __restrict__ dx,
    const float* __restrict__ g, const float* __restrict__ bta,
    float* __restrict__ xo, bf16* __restrict__ xb) {
  __shared__ float red[256];
  int row = blockIdx.x, t = threadIdx.x;
  float s = x[(size_t)row * DM + t] + dx[(size_t)row * DM + t];
  red[t] = s; __syncthreads();
  for (int k = 128; k > 0; k >>= 1) { if (t < k) red[t] += red[t + k]; __syncthreads(); }
  float mean = red[0] * (1.f / DM); __syncthreads();
  float d = s - mean;
  red[t] = d * d; __syncthreads();
  for (int k = 128; k > 0; k >>= 1) { if (t < k) red[t] += red[t + k]; __syncthreads(); }
  float var = red[0] * (1.f / DM);
  float y = d * rsqrtf(var + 1e-5f) * g[t] + bta[t];
  xo[(size_t)row * DM + t] = y;
  xb[(size_t)row * DM + t] = (bf16)y;
}

// =================================================================
// final: LN(x[b, L-1]) with nf, dot proj_W + proj_b -> out[b] (4 floats)
// =================================================================
__global__ __launch_bounds__(256) void inf_final(
    const float* __restrict__ x, const float* __restrict__ g,
    const float* __restrict__ bta, const float* __restrict__ pw,
    const float* __restrict__ pb, float* __restrict__ out) {
  __shared__ float red[256];
  int b = blockIdx.x, t = threadIdx.x;
  size_t row = (size_t)b * LL + (LL - 1);
  float s = x[row * DM + t];
  red[t] = s; __syncthreads();
  for (int k = 128; k > 0; k >>= 1) { if (t < k) red[t] += red[t + k]; __syncthreads(); }
  float mean = red[0] * (1.f / DM); __syncthreads();
  float d = s - mean;
  red[t] = d * d; __syncthreads();
  for (int k = 128; k > 0; k >>= 1) { if (t < k) red[t] += red[t + k]; __syncthreads(); }
  float var = red[0] * (1.f / DM); __syncthreads();
  float y = d * rsqrtf(var + 1e-5f) * g[t] + bta[t];
  red[t] = y * pw[t]; __syncthreads();
  for (int k = 128; k > 0; k >>= 1) { if (t < k) red[t] += red[t + k]; __syncthreads(); }
  if (t == 0) out[b] = red[0] + pb[0];
}

// =================================================================
// host orchestration
// =================================================================
extern "C" void kernel_launch(void* const* d_in, const int* in_sizes, int n_in,
                              void* d_out, int out_size, void* d_ws, size_t ws_size,
                              hipStream_t stream) {
  const float* x_enc = (const float*)d_in[0];
  const int*   samp  = (const int*)d_in[1];
  const float* emb_W = (const float*)d_in[2];
  const float* emb_b = (const float*)d_in[3];
  const float* Wq    = (const float*)d_in[4];
  const float* bq    = (const float*)d_in[5];
  const float* Wk    = (const float*)d_in[6];
  const float* bk    = (const float*)d_in[7];
  const float* Wv    = (const float*)d_in[8];
  const float* bv    = (const float*)d_in[9];
  const float* Wo    = (const float*)d_in[10];
  const float* bo    = (const float*)d_in[11];
  const float* c1W   = (const float*)d_in[12];
  const float* c1b   = (const float*)d_in[13];
  const float* c2W   = (const float*)d_in[14];
  const float* c2b   = (const float*)d_in[15];
  const float* ln1g  = (const float*)d_in[16];
  const float* ln1b  = (const float*)d_in[17];
  const float* ln2g  = (const float*)d_in[18];
  const float* ln2b  = (const float*)d_in[19];
  const float* nfg   = (const float*)d_in[20];
  const float* nfb   = (const float*)d_in[21];
  const float* projW = (const float*)d_in[22];
  const float* projB = (const float*)d_in[23];
  float* out = (float*)d_out;

  // ---- workspace bump allocator ----
  char* ws = (char*)d_ws;
  size_t off = 0;
  auto alloc = [&](size_t bytes) -> char* {
    char* p = ws + off;
    off = (off + bytes + 255) & ~(size_t)255;
    return p;
  };
  const size_t ML = (size_t)BB * LL;              // 8192 rows
  bf16*  xeb  = (bf16*)alloc(ML * ENCIN * 2);
  bf16*  embT = (bf16*)alloc((size_t)DM * ENCIN * 2);
  float* x    = (float*)alloc(ML * DM * 4);
  bf16*  xb   = (bf16*)alloc(ML * DM * 2);
  bf16*  qb   = (bf16*)alloc(ML * HD * 2);        // also reused as a_flat
  bf16*  kb   = (bf16*)alloc(ML * HD * 2);
  bf16*  vb   = (bf16*)alloc(ML * HD * 2);
  float* Mb   = (float*)alloc((size_t)BB * HH * LL * 4);
  int*   mtop = (int*)alloc((size_t)BB * HH * NTOP * 4);
  float* ctxb = (float*)alloc((size_t)BB * HH * NTOP * DM * 4);
  float* newx = (float*)alloc(ML * DM * 4);
  bf16*  y1b  = (bf16*)alloc(ML * DFF * 2);
  bf16*  wtT  = (bf16*)alloc((size_t)HD * DM * 2); // largest weight, reused

  dim3 blk256(256);
  dim3 tblk(32, 8);
  auto tgrid = [](int K, int N) { return dim3((N + 31) / 32, (K + 31) / 32); };

  // templated GEMM dispatch: (f32 out?, bf16 out?, gelu?)
  auto gemm_fb = [&](const bf16* A, const bf16* Bt, const float* bias,
                     float* Cf, bf16* Cb, int M, int N, int K) {
    inf_gemm_bf16<0, true, true><<<dim3(N / 128, M / 128), blk256, 0, stream>>>(
        A, Bt, bias, Cf, Cb, M, N, K);
  };
  auto gemm_b = [&](const bf16* A, const bf16* Bt, const float* bias,
                    bf16* Cb, int M, int N, int K) {
    inf_gemm_bf16<0, false, true><<<dim3(N / 128, M / 128), blk256, 0, stream>>>(
        A, Bt, bias, nullptr, Cb, M, N, K);
  };
  auto gemm_b_gelu = [&](const bf16* A, const bf16* Bt, const float* bias,
                         bf16* Cb, int M, int N, int K) {
    inf_gemm_bf16<1, false, true><<<dim3(N / 128, M / 128), blk256, 0, stream>>>(
        A, Bt, bias, nullptr, Cb, M, N, K);
  };
  auto gemm_f = [&](const bf16* A, const bf16* Bt, const float* bias,
                    float* Cf, int M, int N, int K) {
    inf_gemm_bf16<0, true, false><<<dim3(N / 128, M / 128), blk256, 0, stream>>>(
        A, Bt, bias, Cf, nullptr, M, N, K);
  };

  // ---- embedding: x = x_enc @ emb_W + emb_b ----
  inf_cvt_bf16<<<(int)((ML * ENCIN + 255) / 256), blk256, 0, stream>>>(x_enc, xeb, (int)(ML * ENCIN));
  inf_transpose_cvt<<<tgrid(ENCIN, DM), tblk, 0, stream>>>(emb_W, embT, ENCIN, DM);
  gemm_fb(xeb, embT, emb_b, x, xb, (int)ML, DM, ENCIN);

  for (int l = 0; l < ELAY; ++l) {
    const float* wq = Wq + (size_t)l * DM * HD;
    const float* wk = Wk + (size_t)l * DM * HD;
    const float* wv = Wv + (size_t)l * DM * HD;
    const float* wo = Wo + (size_t)l * HD * DM;

    // Q/K/V projections (bf16 outputs)
    inf_transpose_cvt<<<tgrid(DM, HD), tblk, 0, stream>>>(wq, wtT, DM, HD);
    gemm_b(xb, wtT, bq + (size_t)l * HD, qb, (int)ML, HD, DM);
    inf_transpose_cvt<<<tgrid(DM, HD), tblk, 0, stream>>>(wk, wtT, DM, HD);
    gemm_b(xb, wtT, bk + (size_t)l * HD, kb, (int)ML, HD, DM);
    inf_transpose_cvt<<<tgrid(DM, HD), tblk, 0, stream>>>(wv, wtT, DM, HD);
    gemm_b(xb, wtT, bv + (size_t)l * HD, vb, (int)ML, HD, DM);

    // ProbSparse attention
    inf_sample_m<<<BB * HH * LL / 8, blk256, 0, stream>>>(qb, kb, samp, Mb);
    inf_topk<<<BB * HH, blk256, 0, stream>>>(Mb, mtop);
    inf_ctx<<<dim3(NTOP, HH, BB), blk256, 0, stream>>>(qb, kb, vb, mtop, ctxb);

    // a_flat (reuse qb): zero + scatter, then output projection
    bf16* aflat = qb;
    inf_zero16<<<(int)((ML * HD * 2 / 16 + 255) / 256), blk256, 0, stream>>>(
        (uint4*)aflat, (int)(ML * HD * 2 / 16));
    inf_scatter<<<dim3(NTOP, HH, BB), blk256, 0, stream>>>(ctxb, mtop, aflat);
    inf_transpose_cvt<<<tgrid(HD, DM), tblk, 0, stream>>>(wo, wtT, HD, DM);
    gemm_f(aflat, wtT, bo + (size_t)l * DM, newx, (int)ML, DM, HD);
    inf_ln_res<<<(int)ML, blk256, 0, stream>>>(
        x, newx, ln1g + (size_t)l * DM, ln1b + (size_t)l * DM, x, xb);

    // FFN
    inf_transpose_cvt<<<tgrid(DM, DFF), tblk, 0, stream>>>(c1W + (size_t)l * DM * DFF, wtT, DM, DFF);
    gemm_b_gelu(xb, wtT, c1b + (size_t)l * DFF, y1b, (int)ML, DFF, DM);
    inf_transpose_cvt<<<tgrid(DFF, DM), tblk, 0, stream>>>(c2W + (size_t)l * DFF * DM, wtT, DFF, DM);
    gemm_f(y1b, wtT, c2b + (size_t)l * DM, newx, (int)ML, DM, DFF);
    inf_ln_res<<<(int)ML, blk256, 0, stream>>>(
        x, newx, ln2g + (size_t)l * DM, ln2b + (size_t)l * DM, x, xb);
  }

  inf_final<<<BB, blk256, 0, stream>>>(x, nfg, nfb, projW, projB, out);
}
